// GCN_87986700026523
// MI455X (gfx1250) — matrix operand order repaired and checked by reading, not compiled, not run on hardware
//
#include <hip/hip_runtime.h>
#include <hip/hip_bf16.h>

// ---------------------------------------------------------------------------
// 2-layer GCN for MI455X (gfx1250, wave32).
//   deg -> rsqrt -> [GEMM(WMMA f16 split-precision) -> self+bias -> edge
//   scatter-add -> ReLU] x2
// GEMM uses v_wmma_f32_16x16x32_f16 with split hi/lo f16 operands for
// near-fp32 accuracy. Aggregation is coalesced 64-lane-per-row atomicAdd.
// ---------------------------------------------------------------------------

typedef _Float16 v16h __attribute__((ext_vector_type(16)));
typedef _Float16 v8h  __attribute__((ext_vector_type(8)));
typedef float    v8f  __attribute__((ext_vector_type(8)));

#define D 64   // feature dim for both layers

// ---- degree / norm prep ----------------------------------------------------
__global__ void k_fill1(float* deg, int n) {
    int i = blockIdx.x * blockDim.x + threadIdx.x;
    if (i < n) deg[i] = 1.0f;                       // self-loop
}

__global__ void k_count_deg(const long long* dst, float* deg, int e) {
    int i = blockIdx.x * blockDim.x + threadIdx.x;
    if (i < e) atomicAdd(&deg[(int)dst[i]], 1.0f);
}

__global__ void k_rsqrt(float* d, int n) {
    int i = blockIdx.x * blockDim.x + threadIdx.x;
    if (i < n) d[i] = rsqrtf(d[i]);                 // deg >= 1 always
}

// ---- split fp32 -> (hi,lo) f16 pair (optionally fused ReLU) ---------------
__global__ void k_split_f16(const float* __restrict__ src,
                            _Float16* __restrict__ hi,
                            _Float16* __restrict__ lo,
                            int n, int do_relu) {
    int i = blockIdx.x * blockDim.x + threadIdx.x;
    if (i >= n) return;
    float v = src[i];
    if (do_relu) v = fmaxf(v, 0.0f);
    _Float16 h = (_Float16)v;
    hi[i] = h;
    lo[i] = (_Float16)(v - (float)h);
}

// ---- pack W[64][64] into per-lane WMMA B fragments (hi/lo) ----------------
// Fragment f = kc*4+nt (kc: K chunk of 32, nt: N tile of 16).
// Lane L holds column N = nt*16 + (L&15); element e -> K = kc*32 + (L<16?0:8) + (e<8 ? e : 8+e).
// Stored fragment-major, lane-major: idx = f*512 + lane*16 + e  (contiguous v16h/lane).
__global__ void k_pack_w(const float* __restrict__ W,
                         _Float16* __restrict__ wh,
                         _Float16* __restrict__ wl) {
    int idx = blockIdx.x * blockDim.x + threadIdx.x;   // 0..4095
    if (idx >= 4096) return;
    int f    = idx >> 9;
    int lane = (idx >> 4) & 31;
    int e    = idx & 15;
    int kc   = f >> 2;
    int nt   = f & 3;
    int K    = kc * 32 + ((lane & 16) ? 8 : 0) + (e < 8 ? e : 8 + e);
    int Nc   = nt * 16 + (lane & 15);
    float v  = W[K * D + Nc];
    _Float16 h = (_Float16)v;
    wh[idx] = h;
    wl[idx] = (_Float16)(v - (float)h);
}

// ---- WMMA GEMM: out[16*ntiles][64] = A[.., 64] @ W(packed) ----------------
// One wave per 16-row strip; 24 v_wmma per wave (hi*hi + lo*hi + hi*lo, 2 K
// chunks, 4 N tiles).
__global__ void k_gemm_wmma(const _Float16* __restrict__ ahi,
                            const _Float16* __restrict__ alo,
                            const _Float16* __restrict__ wh,
                            const _Float16* __restrict__ wl,
                            float* __restrict__ out, int ntiles) {
    int wave = (int)((blockIdx.x * blockDim.x + threadIdx.x) >> 5);
    int lane = threadIdx.x & 31;
    if (wave >= ntiles) return;        // uniform per wave: EXEC stays all-ones
    int m0 = wave << 4;
    int m  = m0 + (lane & 15);
    int kb = (lane & 16) ? 8 : 0;      // A half-wave K base within chunk

    const _Float16* ph = ahi + (size_t)m * D;
    const _Float16* pl = alo + (size_t)m * D;

    v8f acc[4];
#pragma unroll
    for (int t = 0; t < 4; ++t)
#pragma unroll
        for (int r = 0; r < 8; ++r) acc[t][r] = 0.0f;

#pragma unroll
    for (int kc = 0; kc < 2; ++kc) {
        v8h h0 = *(const v8h*)(ph + kc * 32 + kb);
        v8h h1 = *(const v8h*)(ph + kc * 32 + 16 + kb);
        v8h l0 = *(const v8h*)(pl + kc * 32 + kb);
        v8h l1 = *(const v8h*)(pl + kc * 32 + 16 + kb);
        v16h Ah, Al;
#pragma unroll
        for (int i = 0; i < 8; ++i) {
            Ah[i] = h0[i]; Ah[8 + i] = h1[i];
            Al[i] = l0[i]; Al[8 + i] = l1[i];
        }
#pragma unroll
        for (int nt = 0; nt < 4; ++nt) {
            int f = kc * 4 + nt;
            v16h Bh = *(const v16h*)(wh + ((size_t)f * 32 + lane) * 16);
            v16h Bl = *(const v16h*)(wl + ((size_t)f * 32 + lane) * 16);
            acc[nt] = __builtin_amdgcn_wmma_f32_16x16x32_f16(
                false, Ah, false, Bh, (short)0, acc[nt], false, false);
            acc[nt] = __builtin_amdgcn_wmma_f32_16x16x32_f16(
                false, Al, false, Bh, (short)0, acc[nt], false, false);
            acc[nt] = __builtin_amdgcn_wmma_f32_16x16x32_f16(
                false, Ah, false, Bl, (short)0, acc[nt], false, false);
        }
    }

    // C/D layout: VGPR r -> row m0 + r + (lane<16?0:8), col nt*16 + (lane&15)
    int rb = (lane & 16) ? 8 : 0;
    int cb = lane & 15;
#pragma unroll
    for (int nt = 0; nt < 4; ++nt)
#pragma unroll
        for (int r = 0; r < 8; ++r)
            out[(size_t)(m0 + rb + r) * D + nt * 16 + cb] = acc[nt][r];
}

// ---- self-loop + bias init: acc[i,c] = h[i,c]*dinv[i]^2 + b[c] ------------
__global__ void k_self_bias(const float* __restrict__ h,
                            const float* __restrict__ dinv,
                            const float* __restrict__ b,
                            float* __restrict__ acc, int n) {
    int i = blockIdx.x * blockDim.x + threadIdx.x;
    if (i >= n) return;
    int node = i >> 6;
    int c    = i & 63;
    float dv = dinv[node];
    acc[i] = h[i] * dv * dv + b[c];
}

// ---- edge scatter: acc[dst] += h[src] * dinv[src]*dinv[dst] ---------------
// 64 consecutive lanes handle one edge row -> fully coalesced 256B gathers
// and atomic-add bursts.
__global__ void k_scatter(const long long* __restrict__ src,
                          const long long* __restrict__ dst,
                          const float* __restrict__ dinv,
                          const float* __restrict__ h,
                          float* __restrict__ acc, int e) {
    int t = blockIdx.x * blockDim.x + threadIdx.x;
    int edge = t >> 6;
    int c    = t & 63;
    if (edge >= e) return;
    int s = (int)src[edge];
    int d = (int)dst[edge];
    __builtin_prefetch(h + (size_t)s * D, 0, 1);   // global_prefetch path
    float nrm = dinv[s] * dinv[d];
    atomicAdd(acc + (size_t)d * D + c, h[(size_t)s * D + c] * nrm);
}

// ---------------------------------------------------------------------------
static inline size_t alignup(size_t x) { return (x + 255) & ~(size_t)255; }

extern "C" void kernel_launch(void* const* d_in, const int* in_sizes, int n_in,
                              void* d_out, int out_size, void* d_ws, size_t ws_size,
                              hipStream_t stream) {
    const float*     x   = (const float*)d_in[0];
    const float*     W1  = (const float*)d_in[1];
    const float*     b1  = (const float*)d_in[2];
    const float*     W2  = (const float*)d_in[3];
    const float*     b2  = (const float*)d_in[4];
    const long long* ei  = (const long long*)d_in[5];

    const int N  = in_sizes[0] / D;
    const int E  = in_sizes[5] / 2;
    const int Np = (N + 15) & ~15;                 // pad to whole 16-row tiles
    const int ntiles = Np / 16;
    const long long* esrc = ei;
    const long long* edst = ei + E;

    // workspace carve-up
    char* ws = (char*)d_ws;
    size_t off = 0;
    float*    dinv = (float*)(ws + off);    off = alignup(off + (size_t)N * 4);
    _Float16* ahi  = (_Float16*)(ws + off); off = alignup(off + (size_t)Np * D * 2);
    _Float16* alo  = (_Float16*)(ws + off); off = alignup(off + (size_t)Np * D * 2);
    float*    hbuf = (float*)(ws + off);    off = alignup(off + (size_t)Np * D * 4);
    float*    abuf = (float*)(ws + off);    off = alignup(off + (size_t)N  * D * 4);
    _Float16* w1h  = (_Float16*)(ws + off); off = alignup(off + 4096 * 2);
    _Float16* w1l  = (_Float16*)(ws + off); off = alignup(off + 4096 * 2);
    _Float16* w2h  = (_Float16*)(ws + off); off = alignup(off + 4096 * 2);
    _Float16* w2l  = (_Float16*)(ws + off); off = alignup(off + 4096 * 2);
    (void)ws_size; (void)n_in;

    const int T = 256;
    const int ND = N * D;
    float* outp = (float*)d_out;

    // --- norm prep ---
    k_fill1    <<<(N + T - 1) / T, T, 0, stream>>>(dinv, N);
    k_count_deg<<<(E + T - 1) / T, T, 0, stream>>>(edst, dinv, E);
    k_rsqrt    <<<(N + T - 1) / T, T, 0, stream>>>(dinv, N);

    // --- pack weights into WMMA B fragments ---
    k_pack_w<<<16, T, 0, stream>>>(W1, w1h, w1l);
    k_pack_w<<<16, T, 0, stream>>>(W2, w2h, w2l);

    // --- layer 1 ---
    k_split_f16<<<(ND + T - 1) / T, T, 0, stream>>>(x, ahi, alo, ND, 0);
    k_gemm_wmma<<<(ntiles + 7) / 8, T, 0, stream>>>(ahi, alo, w1h, w1l, hbuf, ntiles);
    k_self_bias<<<(ND + T - 1) / T, T, 0, stream>>>(hbuf, dinv, b1, abuf, ND);
    {
        long long work = (long long)E * D;
        k_scatter<<<(int)((work + T - 1) / T), T, 0, stream>>>(esrc, edst, dinv, hbuf, abuf, E);
    }

    // --- layer 2 (ReLU fused into the split) ---
    k_split_f16<<<(ND + T - 1) / T, T, 0, stream>>>(abuf, ahi, alo, ND, 1);
    k_gemm_wmma<<<(ntiles + 7) / 8, T, 0, stream>>>(ahi, alo, w2h, w2l, hbuf, ntiles);
    k_self_bias<<<(ND + T - 1) / T, T, 0, stream>>>(hbuf, dinv, b2, outp, ND);
    {
        long long work = (long long)E * D;
        k_scatter<<<(int)((work + T - 1) / T), T, 0, stream>>>(esrc, edst, dinv, hbuf, outp, E);
    }
}
